// MulitAxisSelfAttention_81673098101694
// MI455X (gfx1250) — matrix operand order repaired and checked by reading, not compiled, not run on hardware
//
#include <hip/hip_runtime.h>

typedef __attribute__((ext_vector_type(16))) __bf16 v16bf;
typedef __attribute__((ext_vector_type(8)))  __bf16 v8bf;
typedef __attribute__((ext_vector_type(8)))  float  v8f;

static constexpr float ATTN_SCALE = 0.35355339059327373f; // 8^-0.5 (n_head^-0.5)

// ---- LDS pool layout (bytes). Peak 200704 B < 320KB/WGP. ----
// xp is CHANNEL-MAJOR f32 [256][68] (pad 68 -> +4 bank offset per row, and
// 4 consecutive tokens are 16B-contiguous for b128 LDS ops).
static constexpr int XP_OFF   = 0;         // f32 [256][68]
static constexpr int LNX_OFF  = 69632;     // bf16 [64][256]
static constexpr int Q_OFF    = 102400;    // bf16 [64][256]
static constexpr int K_OFF    = 135168;    // bf16 [64][256]
static constexpr int VT_OFF   = 167936;    // bf16 [256][64]  (v transposed)
static constexpr int SMEM_BYTES = 200704;
static constexpr int P_OFF    = LNX_OFF;   // bf16 [8][64][64] aliases lnx+q (barrier-sequenced)
static constexpr int AO_OFF   = K_OFF;     // bf16 [64][256]   aliases k   (barrier-sequenced)
static constexpr int RED_OFF  = Q_OFF;     // f32 scratch for LN reductions (before q written)

// A fragment (16x32 bf16), ISA 7.12.2: lane<16 row=lane, K={k0..k0+7, k0+16..k0+23};
// lane>=16 same row, K shifted by +8. Two contiguous 16B chunks per lane.
__device__ __forceinline__ v16bf load_a(const __bf16* base, int ld, int row0, int k0, int lane) {
  const __bf16* r = base + (row0 + (lane & 15)) * ld + k0 + ((lane >> 4) << 3);
  v8bf lo = *(const v8bf*)r;
  v8bf hi = *(const v8bf*)(r + 16);
  v16bf a;
#pragma unroll
  for (int e = 0; e < 8; ++e) { a[e] = lo[e]; a[8 + e] = hi[e]; }
  return a;
}

// B fragment (32x16 bf16) from Bt stored [N][K] row-major: lane n=lane&15,
// K = {k0..k0+7, k0+16..k0+23} (+8 for lane>=16), matching load_a.
__device__ __forceinline__ v16bf load_bt(const __bf16* base, int ld, int col0, int k0, int lane) {
  const __bf16* r = base + (col0 + (lane & 15)) * ld + k0 + ((lane >> 4) << 3);
  v8bf lo = *(const v8bf*)r;
  v8bf hi = *(const v8bf*)(r + 16);
  v16bf b;
#pragma unroll
  for (int e = 0; e < 8; ++e) { b[e] = lo[e]; b[8 + e] = hi[e]; }
  return b;
}

__device__ __forceinline__ v8f wmma_bf16(v16bf a, v16bf b, v8f c) {
  return __builtin_amdgcn_wmma_f32_16x16x32_bf16(false, a, false, b, (short)0, c, false, false);
}

// Convert + transpose fp32 weight (K,O) -> bf16 (O,K)
__global__ void convert_T_kernel(const float* __restrict__ w, __bf16* __restrict__ wt,
                                 int K, int O) {
  int idx = blockIdx.x * blockDim.x + threadIdx.x;
  if (idx < K * O) {
    int o = idx / K;
    int k = idx - o * K;
    wt[idx] = (__bf16)w[k * O + o];
  }
}

// MODE 0: window partition (pixel = (gh*8+i, gw*8+j))
// MODE 1: grid partition   (pixel = (i*32+gh, j*32+gw))
template <int MODE>
__global__ __launch_bounds__(256) void maxvit_attn_kernel(
    const float* xin, float* out,
    const float* __restrict__ lng, const float* __restrict__ lnb,
    const __bf16* __restrict__ qkv_wT, const float* __restrict__ qkv_b,
    const __bf16* __restrict__ proj_wT, const float* __restrict__ proj_b,
    const float* __restrict__ btab) {
  extern __shared__ char smem[];
  float*  xp  = (float*)(smem + XP_OFF);     // [ch][68] channel-major
  __bf16* lnx = (__bf16*)(smem + LNX_OFF);
  __bf16* qb  = (__bf16*)(smem + Q_OFF);
  __bf16* kb  = (__bf16*)(smem + K_OFF);
  __bf16* vt  = (__bf16*)(smem + VT_OFF);
  __bf16* pb  = (__bf16*)(smem + P_OFF);
  __bf16* ao  = (__bf16*)(smem + AO_OFF);
  float*  red = (float*)(smem + RED_OFF);

  const int tid  = threadIdx.x;
  const int lane = tid & 31;
  const int w    = tid >> 5;   // wave id 0..7

  const int g  = blockIdx.x;
  const int b  = g >> 10;
  const int gh = (g >> 5) & 31;
  const int gw = g & 31;

  // ---- Phase 0: stage window (64 tok x 256 ch) f32 into LDS (channel-major) ----
  if (MODE == 0) {
    // window rows are 8 contiguous pw floats -> float4 loads AND float4 LDS stores
#pragma unroll 4
    for (int it = 0; it < 16; ++it) {
      int e  = it * 256 + tid;          // [0, 4096) float4 slots
      int ch = e >> 4;
      int r  = e & 15;
      int i  = r >> 1;
      int jq = (r & 1) * 4;
      const float4 v = *(const float4*)&xin[
          (((size_t)b * 256 + ch) * 256 + gh * 8 + i) * 256 + gw * 8 + jq];
      *(float4*)&xp[ch * 68 + i * 8 + jq] = v;
    }
  } else {
#pragma unroll 4
    for (int it = 0; it < 64; ++it) {
      int e = it * 256 + tid;
      int ch = e >> 6, n = e & 63;
      int i = n >> 3, j = n & 7;
      xp[ch * 68 + n] =
          xin[(((size_t)b * 256 + ch) * 256 + i * 32 + gh) * 256 + j * 32 + gw];
    }
  }
  __syncthreads();

  // ---- Phase 1: LayerNorm -> lnx bf16 ----
  {
    int tok = tid & 63, part = tid >> 6, c0 = part * 64;
    float s = 0.f, s2 = 0.f;
    for (int c = 0; c < 64; ++c) { float v = xp[(c0 + c) * 68 + tok]; s += v; s2 += v * v; }
    red[(part * 64 + tok) * 2]     = s;
    red[(part * 64 + tok) * 2 + 1] = s2;
    __syncthreads();
    float* stats = red + 512;
    if (tid < 64) {
      float S = 0.f, S2 = 0.f;
      for (int p2 = 0; p2 < 4; ++p2) { S += red[(p2 * 64 + tid) * 2]; S2 += red[(p2 * 64 + tid) * 2 + 1]; }
      float mu  = S * (1.f / 256.f);
      float var = S2 * (1.f / 256.f) - mu * mu;
      stats[tid * 2]     = mu;
      stats[tid * 2 + 1] = rsqrtf(var + 1e-5f);
    }
    __syncthreads();
    float mu = stats[tok * 2], rs = stats[tok * 2 + 1];
    for (int c = 0; c < 64; ++c) {
      int cc = c0 + c;
      lnx[tok * 256 + cc] = (__bf16)((xp[cc * 68 + tok] - mu) * rs * lng[cc] + lnb[cc]);
    }
  }
  __syncthreads();

  // ---- Phase 2: QKV = lnx(64x256) @ Wqkv(256x768) + b ----
  // Wave w owns N-tiles {w*2+t}, {16+w*2+t}, {32+w*2+t}: destination region is
  // compile-time known per sub-loop -> no divergent stores.
  for (int mt = 0; mt < 4; ++mt) {
    v16bf afr[8];
#pragma unroll
    for (int ks = 0; ks < 8; ++ks) afr[ks] = load_a(lnx, 256, mt * 16, ks * 32, lane);

    auto gemm = [&](int nt) {
      v8f acc = {};
#pragma unroll
      for (int ks = 0; ks < 8; ++ks)
        acc = wmma_bf16(afr[ks], load_bt(qkv_wT, 256, nt * 16, ks * 32, lane), acc);
      return acc;
    };

    const int tok0 = mt * 16 + ((lane >> 4) << 3);  // 8 consecutive D rows
    for (int t = 0; t < 2; ++t) {
      { // Q tile
        int nt = w * 2 + t;
        v8f acc = gemm(nt);
        int col = nt * 16 + (lane & 15);
        float bias = qkv_b[col];
#pragma unroll
        for (int p = 0; p < 8; ++p)
          qb[(tok0 + p) * 256 + col] = (__bf16)(acc[p] + bias);
      }
      { // K tile
        int nt = 16 + w * 2 + t;
        v8f acc = gemm(nt);
        int col = nt * 16 + (lane & 15);       // [256,512)
        float bias = qkv_b[col];
        int kc = col - 256;
#pragma unroll
        for (int p = 0; p < 8; ++p)
          kb[(tok0 + p) * 256 + kc] = (__bf16)(acc[p] + bias);
      }
      { // V tile (transposed layout: 8 consecutive tokens -> one b128 store)
        int nt = 32 + w * 2 + t;
        v8f acc = gemm(nt);
        int col = nt * 16 + (lane & 15);       // [512,768)
        float bias = qkv_b[col];
        int vc = col - 512;
        v8bf pk;
#pragma unroll
        for (int p = 0; p < 8; ++p) pk[p] = (__bf16)(acc[p] + bias);
        *(v8bf*)&vt[vc * 64 + tok0] = pk;
      }
    }
  }
  __syncthreads();

  // ---- Phase 3: wave w = head h: S = q_h k_h^T, softmax in registers ----
  const int h = w;
  v16bf aq[4], bk[4];
#pragma unroll
  for (int mt = 0; mt < 4; ++mt) aq[mt] = load_a(qb, 256, mt * 16, h * 32, lane);
#pragma unroll
  for (int nt = 0; nt < 4; ++nt) bk[nt] = load_bt(kb, 256, nt * 16, h * 32, lane);
  __syncthreads();  // all q/k reads done before pb (aliases lnx+q) is written

  for (int mt = 0; mt < 4; ++mt) {
    v8f srow[4];
#pragma unroll
    for (int nt = 0; nt < 4; ++nt) { v8f z = {}; srow[nt] = wmma_bf16(aq[mt], bk[nt], z); }
#pragma unroll
    for (int p = 0; p < 8; ++p) {
      int r  = mt * 16 + p + ((lane >> 4) << 3);
      int ir = r >> 3, jr = r & 7;
      float v4[4];
      float m = -3.0e38f;
#pragma unroll
      for (int nt = 0; nt < 4; ++nt) {
        int c = nt * 16 + (lane & 15);
        int ic = c >> 3, jc = c & 7;
        int idx = (ir - ic + 7) * 15 + (jr - jc + 7);
        float val = srow[nt][p] * ATTN_SCALE + btab[idx * 8 + h];
        v4[nt] = val;
        m = fmaxf(m, val);
      }
#pragma unroll
      for (int msk = 1; msk <= 8; msk <<= 1) m = fmaxf(m, __shfl_xor(m, msk, 32));
      float s = 0.f;
#pragma unroll
      for (int nt = 0; nt < 4; ++nt) { v4[nt] = __expf(v4[nt] - m); s += v4[nt]; }
#pragma unroll
      for (int msk = 1; msk <= 8; msk <<= 1) s += __shfl_xor(s, msk, 32);
      float inv = 1.f / s;
#pragma unroll
      for (int nt = 0; nt < 4; ++nt)
        pb[h * 4096 + r * 64 + nt * 16 + (lane & 15)] = (__bf16)(v4[nt] * inv);
    }
  }

  // ---- Phase 4: O_h = P(64x64) @ V_h(64x32); pb is same-wave data (LDS in-order) ----
  const __bf16* php = pb + h * 4096;
  for (int mt = 0; mt < 4; ++mt) {
    v16bf ap0 = load_a(php, 64, mt * 16, 0, lane);
    v16bf ap1 = load_a(php, 64, mt * 16, 32, lane);
#pragma unroll
    for (int nt = 0; nt < 2; ++nt) {
      v8f acc = {};
      acc = wmma_bf16(ap0, load_bt(vt, 64, h * 32 + nt * 16, 0, lane), acc);
      acc = wmma_bf16(ap1, load_bt(vt, 64, h * 32 + nt * 16, 32, lane), acc);
      int ch = h * 32 + nt * 16 + (lane & 15);
#pragma unroll
      for (int p = 0; p < 8; ++p) {
        int tok = mt * 16 + p + ((lane >> 4) << 3);
        ao[tok * 256 + ch] = (__bf16)acc[p];
      }
    }
  }
  __syncthreads();

  // ---- Phase 5: proj(64x256 @ 256x256) + bias + residual -> (B,C,H,W) ----
  for (int mt = 0; mt < 4; ++mt) {
    v16bf afr[8];
#pragma unroll
    for (int ks = 0; ks < 8; ++ks) afr[ks] = load_a(ao, 256, mt * 16, ks * 32, lane);
#pragma unroll
    for (int t = 0; t < 2; ++t) {
      int nt = w * 2 + t;
      v8f acc = {};
#pragma unroll
      for (int ks = 0; ks < 8; ++ks)
        acc = wmma_bf16(afr[ks], load_bt(proj_wT, 256, nt * 16, ks * 32, lane), acc);
      int ch = nt * 16 + (lane & 15);
      float bias = proj_b[ch];
      // 8 acc rows = tokens tok0..tok0+7 (consecutive): residual via two b128 LDS loads
      const int tok0 = mt * 16 + ((lane >> 4) << 3);
      float4 r0 = *(const float4*)&xp[ch * 68 + tok0];
      float4 r1 = *(const float4*)&xp[ch * 68 + tok0 + 4];
      float res[8];
      res[0] = r0.x + acc[0] + bias; res[1] = r0.y + acc[1] + bias;
      res[2] = r0.z + acc[2] + bias; res[3] = r0.w + acc[3] + bias;
      res[4] = r1.x + acc[4] + bias; res[5] = r1.y + acc[5] + bias;
      res[6] = r1.z + acc[6] + bias; res[7] = r1.w + acc[7] + bias;
      int i = mt * 2 + (lane >> 4);   // token row index within window
      if (MODE == 0) {
        size_t base = (((size_t)b * 256 + ch) * 256 + gh * 8 + i) * 256 + gw * 8;
        float4 lo = {res[0], res[1], res[2], res[3]};
        float4 hi = {res[4], res[5], res[6], res[7]};
        *(float4*)&out[base]     = lo;
        *(float4*)&out[base + 4] = hi;
      } else {
#pragma unroll
        for (int p = 0; p < 8; ++p) {
          out[(((size_t)b * 256 + ch) * 256 + i * 32 + gh) * 256 + p * 32 + gw] = res[p];
        }
      }
    }
  }
}

extern "C" void kernel_launch(void* const* d_in, const int* in_sizes, int n_in,
                              void* d_out, int out_size, void* d_ws, size_t ws_size,
                              hipStream_t stream) {
  const float* x        = (const float*)d_in[0];
  const float* b_ln_g   = (const float*)d_in[1];
  const float* b_ln_b   = (const float*)d_in[2];
  const float* b_qkv_w  = (const float*)d_in[3];
  const float* b_qkv_b  = (const float*)d_in[4];
  const float* b_proj_w = (const float*)d_in[5];
  const float* b_proj_b = (const float*)d_in[6];
  const float* b_btab   = (const float*)d_in[7];
  const float* g_ln_g   = (const float*)d_in[8];
  const float* g_ln_b   = (const float*)d_in[9];
  const float* g_qkv_w  = (const float*)d_in[10];
  const float* g_qkv_b  = (const float*)d_in[11];
  const float* g_proj_w = (const float*)d_in[12];
  const float* g_proj_b = (const float*)d_in[13];
  const float* g_btab   = (const float*)d_in[14];
  float* out = (float*)d_out;

  __bf16* ws      = (__bf16*)d_ws;
  __bf16* b_qkvT  = ws;                              // 768*256
  __bf16* b_projT = b_qkvT + 768 * 256;              // 256*256
  __bf16* g_qkvT  = b_projT + 256 * 256;             // 768*256
  __bf16* g_projT = g_qkvT + 768 * 256;              // 256*256

  convert_T_kernel<<<(768 * 256 + 255) / 256, 256, 0, stream>>>(b_qkv_w, b_qkvT, 256, 768);
  convert_T_kernel<<<(256 * 256 + 255) / 256, 256, 0, stream>>>(b_proj_w, b_projT, 256, 256);
  convert_T_kernel<<<(768 * 256 + 255) / 256, 256, 0, stream>>>(g_qkv_w, g_qkvT, 256, 768);
  convert_T_kernel<<<(256 * 256 + 255) / 256, 256, 0, stream>>>(g_proj_w, g_projT, 256, 256);

  // Block 1: window attention, x -> out
  maxvit_attn_kernel<0><<<2048, 256, SMEM_BYTES, stream>>>(
      x, out, b_ln_g, b_ln_b, b_qkvT, b_qkv_b, b_projT, b_proj_b, b_btab);
  // Block 2: grid attention, out -> out (each WG reads its disjoint window before writing it)
  maxvit_attn_kernel<1><<<2048, 256, SMEM_BYTES, stream>>>(
      out, out, g_ln_g, g_ln_b, g_qkvT, g_qkv_b, g_projT, g_proj_b, g_btab);
}
